// MESGM_15857019256842
// MI455X (gfx1250) — compile-verified, matrix operand order checked by reading, not run on hardware
//
#include <hip/hip_runtime.h>
#include <hip/hip_bf16.h>
#include <math.h>

typedef __bf16 bf16_t;
typedef __attribute__((ext_vector_type(16))) __bf16 v16bf;
typedef __attribute__((ext_vector_type(8)))  __bf16 v8bf;
typedef __attribute__((ext_vector_type(8)))  float  v8f;
typedef __attribute__((ext_vector_type(4)))  int    v4i;

#define B_  16
#define S_  512
#define H_  768
#define M_  32
#define LC_ 32
#define NL_ 7
#define I_  3072
#define NH_ 8
#define DH_ 96   // H/NH

#if defined(__has_builtin)
#if __has_builtin(__builtin_amdgcn_global_load_async_to_lds_b128)
#define USE_ASYNC_LDS 1
typedef __attribute__((address_space(1))) v4i* gas_v4i_p;   // global int4*
typedef __attribute__((address_space(3))) v4i* las_v4i_p;   // LDS int4*
#endif
#endif

// ---------------------------------------------------------------- weights: W[K,N] f32 -> WT[N,K] bf16
__global__ void k_transpose_bf16(const float* __restrict__ W, bf16_t* __restrict__ WT, int K, int N) {
  int i = blockIdx.x * blockDim.x + threadIdx.x;
  if (i >= K * N) return;
  int k = i / N, n = i - k * N;
  WT[(size_t)n * K + k] = (bf16_t)W[i];
}

// ---------------------------------------------------------------- gather clause tokens, mask, -> bf16
__global__ void k_gather(const float* __restrict__ ehs, const int* __restrict__ wr,
                         const int* __restrict__ wrm, bf16_t* __restrict__ X0) {
  int r = blockIdx.x;                 // 0..B*M*LC-1
  int b = r >> 10;                    // / (M*LC)
  int idx = wr[r];
  float mask = (float)wrm[r];
  const float* src = ehs + ((size_t)(b * S_ + idx)) * H_;
  bf16_t* dst = X0 + (size_t)r * H_;
  for (int c = threadIdx.x; c < H_; c += blockDim.x)
    dst[c] = (bf16_t)(src[c] * mask);
}

// ---------------------------------------------------------------- WMMA fragment load (16-bit A layout)
__device__ __forceinline__ v16bf load_frag(const bf16_t* __restrict__ base, unsigned off) {
  v8bf lo = *(const v8bf*)(base + off);
  v8bf hi = *(const v8bf*)(base + off + 16);
  v16bf r;
#pragma unroll
  for (int i = 0; i < 8; ++i) { r[i] = lo[i]; r[i + 8] = hi[i]; }
  return r;
}

#define WMMA8(AA0, AA1, BB0, BB1, BB2, BB3)                                                                        \
  acc[0][0] = __builtin_amdgcn_wmma_f32_16x16x32_bf16(false, AA0, false, BB0, (short)0, acc[0][0], false, false);  \
  acc[0][1] = __builtin_amdgcn_wmma_f32_16x16x32_bf16(false, AA0, false, BB1, (short)0, acc[0][1], false, false);  \
  acc[0][2] = __builtin_amdgcn_wmma_f32_16x16x32_bf16(false, AA0, false, BB2, (short)0, acc[0][2], false, false);  \
  acc[0][3] = __builtin_amdgcn_wmma_f32_16x16x32_bf16(false, AA0, false, BB3, (short)0, acc[0][3], false, false);  \
  acc[1][0] = __builtin_amdgcn_wmma_f32_16x16x32_bf16(false, AA1, false, BB0, (short)0, acc[1][0], false, false);  \
  acc[1][1] = __builtin_amdgcn_wmma_f32_16x16x32_bf16(false, AA1, false, BB1, (short)0, acc[1][1], false, false);  \
  acc[1][2] = __builtin_amdgcn_wmma_f32_16x16x32_bf16(false, AA1, false, BB2, (short)0, acc[1][2], false, false);  \
  acc[1][3] = __builtin_amdgcn_wmma_f32_16x16x32_bf16(false, AA1, false, BB3, (short)0, acc[1][3], false, false)

// ---------------------------------------------------------------- generic bf16 WMMA GEMM
// C[Mr,N] = act(A[Mr,K] @ BT[N,K]^T + bias + resid). One wave -> 32x64 tile
// (2 A frags x 4 B frags = 8 WMMAs per 32-K phase), ping-pong double-buffered
// K-loop (step 64) so there is no register rotation between phases.
// act: 0=none 1=relu 2=gelu(erf).  Requires Mr%32==0, N%64==0, K%64==0.
__global__ __launch_bounds__(128, 1)
void k_gemm_wmma(const bf16_t* __restrict__ A, const bf16_t* __restrict__ BT,
                 const float* __restrict__ bias, const float* __restrict__ resid,
                 float* __restrict__ Cf, bf16_t* __restrict__ Cb,
                 int Mr, int N, int K, int act) {
  int ngrp = N >> 6;                                        // groups of four 16-wide n tiles
  int wave = blockIdx.x * (blockDim.x >> 5) + (threadIdx.x >> 5);
  int total = (Mr >> 5) * ngrp;                             // 32-row m blocks
  if (wave >= total) return;
  int tm = wave / ngrp;
  int tg = wave - tm * ngrp;
  int lane = threadIdx.x & 31;
  int r16 = lane & 15;
  int half = lane >> 4;
  // 16-bit A layout: lanes 0-15 row=lane, dwords 0-3 = K+half*8..+7, dwords 4-7 = K+16+half*8..+7
  unsigned ao0 = (unsigned)((tm * 32 + r16) * K + half * 8);
  unsigned ao1 = ao0 + (unsigned)(16 * K);
  unsigned bo0 = (unsigned)((tg * 64 + 0 * 16 + r16) * K + half * 8);
  unsigned bo1 = (unsigned)((tg * 64 + 1 * 16 + r16) * K + half * 8);
  unsigned bo2 = (unsigned)((tg * 64 + 2 * 16 + r16) * K + half * 8);
  unsigned bo3 = (unsigned)((tg * 64 + 3 * 16 + r16) * K + half * 8);

  v8f acc[2][4];
#pragma unroll
  for (int s = 0; s < 2; ++s)
#pragma unroll
    for (int q = 0; q < 4; ++q) acc[s][q] = (v8f){};

  // ping set (x*): fragments for k0 = 0
  v16bf xa0 = load_frag(A, ao0);
  v16bf xa1 = load_frag(A, ao1);
  v16bf xb0 = load_frag(BT, bo0);
  v16bf xb1 = load_frag(BT, bo1);
  v16bf xb2 = load_frag(BT, bo2);
  v16bf xb3 = load_frag(BT, bo3);

  for (unsigned k0 = 0; k0 < (unsigned)K; k0 += 64) {
    // pong set (y*): fragments for k0+32 -- loads overlap the x-phase WMMAs
    v16bf ya0 = load_frag(A, ao0 + k0 + 32);
    v16bf ya1 = load_frag(A, ao1 + k0 + 32);
    v16bf yb0 = load_frag(BT, bo0 + k0 + 32);
    v16bf yb1 = load_frag(BT, bo1 + k0 + 32);
    v16bf yb2 = load_frag(BT, bo2 + k0 + 32);
    v16bf yb3 = load_frag(BT, bo3 + k0 + 32);
    __builtin_prefetch(A + ao0 + k0 + 512, 0, 3);   // global_prefetch_b8
    WMMA8(xa0, xa1, xb0, xb1, xb2, xb3);
    if (k0 + 64 < (unsigned)K) {
      // refill ping set for k0+64 -- loads overlap the y-phase WMMAs
      xa0 = load_frag(A, ao0 + k0 + 64);
      xa1 = load_frag(A, ao1 + k0 + 64);
      xb0 = load_frag(BT, bo0 + k0 + 64);
      xb1 = load_frag(BT, bo1 + k0 + 64);
      xb2 = load_frag(BT, bo2 + k0 + 64);
      xb3 = load_frag(BT, bo3 + k0 + 64);
    }
    WMMA8(ya0, ya1, yb0, yb1, yb2, yb3);
  }

  // C/D layout: lane holds column n=r16 (+tile offset), rows m = half*8 + v
#pragma unroll
  for (int s = 0; s < 2; ++s) {
#pragma unroll
    for (int q = 0; q < 4; ++q) {
      int n = tg * 64 + q * 16 + r16;
      float bv = bias ? bias[n] : 0.f;
#pragma unroll
      for (int v = 0; v < 8; ++v) {
        int m = tm * 32 + s * 16 + half * 8 + v;
        float x = acc[s][q][v] + bv;
        if (resid) x += resid[(size_t)m * N + n];
        if (act == 1) x = fmaxf(x, 0.f);
        else if (act == 2) x = 0.5f * x * (1.f + erff(x * 0.70710678118f));
        if (Cf) Cf[(size_t)m * N + n] = x;
        if (Cb) Cb[(size_t)m * N + n] = (bf16_t)x;
      }
    }
  }
}

// ---------------------------------------------------------------- Hout = relu(adj @ Y + bias), bf16 out
// adj tile (32x32 f32 = 4KB) staged in LDS via GLOBAL_LOAD_ASYNC_TO_LDS_B128 when available.
__global__ void k_adj_mm(const float* __restrict__ adj, const float* __restrict__ Y,
                         const float* __restrict__ bias, bf16_t* __restrict__ Hout) {
  __shared__ float adj_s[LC_ * LC_];
  int bm = blockIdx.x;                                      // 0..B*M-1
  const float* a = adj + (size_t)bm * LC_ * LC_;
#ifdef USE_ASYNC_LDS
  {
    int t = threadIdx.x;                                    // 256 threads x 16B = 4KB
    __builtin_amdgcn_global_load_async_to_lds_b128(
        (gas_v4i_p)(a + t * 4), (las_v4i_p)(adj_s + t * 4), 0, 0);
    asm volatile("s_wait_asynccnt 0x0" ::: "memory");
  }
#else
  for (int i = threadIdx.x; i < LC_ * LC_; i += blockDim.x) adj_s[i] = a[i];
#endif
  __syncthreads();
  size_t row0 = (size_t)bm * LC_;
  for (int h = threadIdx.x; h < H_; h += blockDim.x) {
    float y[LC_];
#pragma unroll
    for (int j = 0; j < LC_; ++j) y[j] = Y[(row0 + j) * H_ + h];
    float bv = bias[h];
    for (int i = 0; i < LC_; ++i) {
      float s = bv;
#pragma unroll
      for (int j = 0; j < LC_; ++j) s += adj_s[i * LC_ + j] * y[j];
      Hout[(row0 + i) * H_ + h] = (bf16_t)fmaxf(s, 0.f);
    }
  }
}

// ---------------------------------------------------------------- masked max+mean pool over LC, -> [B*M, 4H] bf16
__global__ void k_pool(const bf16_t* __restrict__ X0, const bf16_t* __restrict__ H2,
                       const int* __restrict__ wrm, bf16_t* __restrict__ pooled) {
  __shared__ float msk[LC_];
  __shared__ float slen;
  int bm = blockIdx.x;
  if (threadIdx.x < LC_) msk[threadIdx.x] = (float)wrm[bm * LC_ + threadIdx.x];
  __syncthreads();
  if (threadIdx.x == 0) {
    float s = 0.f;
    for (int l = 0; l < LC_; ++l) s += msk[l];
    slen = s + 1e-45f;
  }
  __syncthreads();
  size_t row0 = (size_t)bm * LC_;
  for (int c = threadIdx.x; c < 2 * H_; c += blockDim.x) {
    const bf16_t* base = (c < H_) ? (X0 + row0 * H_ + c) : (H2 + row0 * H_ + (c - H_));
    float mx = -INFINITY, sm = 0.f;
    for (int l = 0; l < LC_; ++l) {
      float v = (float)base[(size_t)l * H_] * msk[l];
      mx = fmaxf(mx, v);
      sm += v;
    }
    pooled[(size_t)bm * (4 * H_) + c] = (bf16_t)mx;
    pooled[(size_t)bm * (4 * H_) + 2 * H_ + c] = (bf16_t)(sm / slen);
  }
}

// ---------------------------------------------------------------- attention core, one wave per (b, head)
__global__ void k_attn(const float* __restrict__ Q, const float* __restrict__ Kt,
                       const float* __restrict__ V, const int* __restrict__ cnm,
                       bf16_t* __restrict__ ctx) {
  int bh = blockIdx.x;
  int b = bh >> 3;
  int hh = bh & 7;
  int i = threadIdx.x;                                      // query index 0..31
  size_t base = ((size_t)b * M_) * H_ + (size_t)hh * DH_;
  float q[DH_];
#pragma unroll
  for (int d = 0; d < DH_; ++d) q[d] = Q[base + (size_t)i * H_ + d];
  float sc[M_];
  float mx = -INFINITY;
  for (int j = 0; j < M_; ++j) {
    const float* kr = Kt + base + (size_t)j * H_;
    float s = 0.f;
#pragma unroll
    for (int d = 0; d < DH_; ++d) s += q[d] * kr[d];
    s = s * 0.1020620726f + (1.f - (float)cnm[b * M_ + j]) * -10000.f;  // 1/sqrt(96)
    sc[j] = s;
    mx = fmaxf(mx, s);
  }
  float sum = 0.f;
  for (int j = 0; j < M_; ++j) { sc[j] = expf(sc[j] - mx); sum += sc[j]; }
  float inv = 1.f / sum;
  for (int d = 0; d < DH_; ++d) {
    float acc = 0.f;
    for (int j = 0; j < M_; ++j) acc += sc[j] * V[base + (size_t)j * H_ + d];
    ctx[base + (size_t)i * H_ + d] = (bf16_t)(acc * inv);
  }
}

// ---------------------------------------------------------------- LayerNorm over H, fp32 + optional bf16 out
__global__ void k_ln(const float* __restrict__ x, const float* __restrict__ g,
                     const float* __restrict__ bta, float* __restrict__ of,
                     bf16_t* __restrict__ ob) {
  __shared__ float s1[256], s2[256];
  int r = blockIdx.x;
  const float* xr = x + (size_t)r * H_;
  float sum = 0.f, sq = 0.f;
  for (int c = threadIdx.x; c < H_; c += blockDim.x) { float v = xr[c]; sum += v; sq += v * v; }
  s1[threadIdx.x] = sum; s2[threadIdx.x] = sq;
  __syncthreads();
  for (int o = 128; o > 0; o >>= 1) {
    if ((int)threadIdx.x < o) { s1[threadIdx.x] += s1[threadIdx.x + o]; s2[threadIdx.x] += s2[threadIdx.x + o]; }
    __syncthreads();
  }
  float mu = s1[0] / (float)H_;
  float var = s2[0] / (float)H_ - mu * mu;
  float rs = rsqrtf(var + 1e-12f);
  for (int c = threadIdx.x; c < H_; c += blockDim.x) {
    float y = (xr[c] - mu) * rs * g[c] + bta[c];
    if (of) of[(size_t)r * H_ + c] = y;
    if (ob) ob[(size_t)r * H_ + c] = (bf16_t)y;
  }
}

// ---------------------------------------------------------------- decoder [512,768]@[768,7]+b (fp32, tiny)
__global__ void k_dec(const float* __restrict__ X, const float* __restrict__ W,
                      const float* __restrict__ bia, float* __restrict__ pred) {
  int t = blockIdx.x * blockDim.x + threadIdx.x;
  if (t >= B_ * M_ * NL_) return;
  int r = t / NL_, c = t - r * NL_;
  float s = bia[c];
  const float* xr = X + (size_t)r * H_;
  for (int k = 0; k < H_; ++k) s += xr[k] * W[k * NL_ + c];
  pred[t] = s;
}

// ---------------------------------------------------------------- masked KL loss -> scalar
__global__ void k_loss(const float* __restrict__ pred, const float* __restrict__ tgt,
                       const int* __restrict__ cnm, float* __restrict__ out) {
  __shared__ float sn[256], sd[256];
  float num = 0.f, den = 0.f;
  for (int r = threadIdx.x; r < B_ * M_; r += blockDim.x) {
    const float* p = pred + r * NL_;
    float mx = p[0];
    for (int c = 1; c < NL_; ++c) mx = fmaxf(mx, p[c]);
    float s = 0.f;
    for (int c = 0; c < NL_; ++c) s += expf(p[c] - mx);
    float lse = logf(s) + mx;
    float kl = 0.f;
    const float* t = tgt + r * NL_;
    for (int c = 0; c < NL_; ++c) {
      float tv = t[c];
      float logp = p[c] - lse;
      kl += (tv > 0.f ? tv * logf(tv) : 0.f) - tv * logp;
    }
    kl *= (1.f / (float)NL_);
    float m = (float)cnm[r];
    num += kl * m;
    den += m;
  }
  sn[threadIdx.x] = num; sd[threadIdx.x] = den;
  __syncthreads();
  for (int o = 128; o > 0; o >>= 1) {
    if ((int)threadIdx.x < o) { sn[threadIdx.x] += sn[threadIdx.x + o]; sd[threadIdx.x] += sd[threadIdx.x + o]; }
    __syncthreads();
  }
  if (threadIdx.x == 0) out[0] = sn[0] / sd[0];
}

// ================================================================ host
extern "C" void kernel_launch(void* const* d_in, const int* in_sizes, int n_in,
                              void* d_out, int out_size, void* d_ws, size_t ws_size,
                              hipStream_t stream) {
  const float* ehs  = (const float*)d_in[0];
  const int*   wr   = (const int*)d_in[1];
  const int*   wrm  = (const int*)d_in[2];
  const int*   cnm  = (const int*)d_in[3];
  const float* adj  = (const float*)d_in[4];
  const float* tgt  = (const float*)d_in[5];
  const float* gc1w = (const float*)d_in[6];  const float* gc1b = (const float*)d_in[7];
  const float* gc2w = (const float*)d_in[8];  const float* gc2b = (const float*)d_in[9];
  const float* prjw = (const float*)d_in[10]; const float* prjb = (const float*)d_in[11];
  const float* qw   = (const float*)d_in[12]; const float* qb   = (const float*)d_in[13];
  const float* kw   = (const float*)d_in[14]; const float* kb   = (const float*)d_in[15];
  const float* vw   = (const float*)d_in[16]; const float* vb   = (const float*)d_in[17];
  const float* aow  = (const float*)d_in[18]; const float* aob  = (const float*)d_in[19];
  const float* ln1g = (const float*)d_in[20]; const float* ln1b = (const float*)d_in[21];
  const float* intw = (const float*)d_in[22]; const float* intb = (const float*)d_in[23];
  const float* outw = (const float*)d_in[24]; const float* outb = (const float*)d_in[25];
  const float* ln2g = (const float*)d_in[26]; const float* ln2b = (const float*)d_in[27];
  const float* decw = (const float*)d_in[28]; const float* decb = (const float*)d_in[29];

  char* ws = (char*)d_ws;
  size_t off = 0;
  auto alloc = [&](size_t bytes) -> void* {
    void* p = ws + off;
    off = (off + bytes + 255) & ~(size_t)255;
    return p;
  };
  const size_t R = (size_t)B_ * M_ * LC_;   // 16384 token rows
  const size_t P = (size_t)B_ * M_;         // 512 clause rows
  bf16_t* X0   = (bf16_t*)alloc(R * H_ * 2);
  float*  Yf   = (float*) alloc(R * H_ * 4);
  bf16_t* H1   = (bf16_t*)alloc(R * H_ * 2);
  bf16_t* H2   = (bf16_t*)alloc(R * H_ * 2);
  bf16_t* gc1T = (bf16_t*)alloc((size_t)H_ * H_ * 2);
  bf16_t* gc2T = (bf16_t*)alloc((size_t)H_ * H_ * 2);
  bf16_t* prjT = (bf16_t*)alloc((size_t)(4 * H_) * H_ * 2);
  bf16_t* qT   = (bf16_t*)alloc((size_t)H_ * H_ * 2);
  bf16_t* kT   = (bf16_t*)alloc((size_t)H_ * H_ * 2);
  bf16_t* vT   = (bf16_t*)alloc((size_t)H_ * H_ * 2);
  bf16_t* aoT  = (bf16_t*)alloc((size_t)H_ * H_ * 2);
  bf16_t* intT = (bf16_t*)alloc((size_t)H_ * I_ * 2);
  bf16_t* outT = (bf16_t*)alloc((size_t)I_ * H_ * 2);
  bf16_t* pool = (bf16_t*)alloc(P * 4 * H_ * 2);
  float*  cvf  = (float*) alloc(P * H_ * 4);
  bf16_t* cvb  = (bf16_t*)alloc(P * H_ * 2);
  float*  qf   = (float*) alloc(P * H_ * 4);
  float*  kf   = (float*) alloc(P * H_ * 4);
  float*  vf   = (float*) alloc(P * H_ * 4);
  bf16_t* ctxb = (bf16_t*)alloc(P * H_ * 2);
  float*  ares = (float*) alloc(P * H_ * 4);   // ctx@ao + b + cv
  float*  aof  = (float*) alloc(P * H_ * 4);   // LN1 out f32
  bf16_t* aob16= (bf16_t*)alloc(P * H_ * 2);   // LN1 out bf16
  bf16_t* intb16=(bf16_t*)alloc(P * I_ * 2);   // gelu out bf16
  float*  fres = (float*) alloc(P * H_ * 4);   // inter@out + b + attn_out
  float*  o2f  = (float*) alloc(P * H_ * 4);   // LN2 out
  float*  pred = (float*) alloc(P * NL_ * 4);

  auto tr = [&](const float* W, bf16_t* WT, int K, int N) {
    int tot = K * N;
    k_transpose_bf16<<<(tot + 255) / 256, 256, 0, stream>>>(W, WT, K, N);
  };
  auto gemm = [&](const bf16_t* A, const bf16_t* BT, const float* bias, const float* resid,
                  float* Cf, bf16_t* Cb, int Mr, int N, int K, int act) {
    int total = (Mr / 32) * (N / 64);
    k_gemm_wmma<<<(total + 3) / 4, 128, 0, stream>>>(A, BT, bias, resid, Cf, Cb, Mr, N, K, act);
  };

  // weights -> transposed bf16
  tr(gc1w, gc1T, H_, H_);
  tr(gc2w, gc2T, H_, H_);
  tr(prjw, prjT, 4 * H_, H_);
  tr(qw, qT, H_, H_);   tr(kw, kT, H_, H_);
  tr(vw, vT, H_, H_);   tr(aow, aoT, H_, H_);
  tr(intw, intT, H_, I_);
  tr(outw, outT, I_, H_);

  // clause gather (+mask) -> bf16
  k_gather<<<(int)R, 256, 0, stream>>>(ehs, wr, wrm, X0);

  // GCN layer 1: Y = X0 @ gc1 ; H1 = relu(adj@Y + b1)
  gemm(X0, gc1T, nullptr, nullptr, Yf, nullptr, (int)R, H_, H_, 0);
  k_adj_mm<<<(int)P, 256, 0, stream>>>(adj, Yf, gc1b, H1);
  // GCN layer 2
  gemm(H1, gc2T, nullptr, nullptr, Yf, nullptr, (int)R, H_, H_, 0);
  k_adj_mm<<<(int)P, 256, 0, stream>>>(adj, Yf, gc2b, H2);

  // pooling -> [512, 4H]; cv = relu(pool @ proj + b)
  k_pool<<<(int)P, 256, 0, stream>>>(X0, H2, wrm, pool);
  gemm(pool, prjT, prjb, nullptr, cvf, cvb, (int)P, H_, 4 * H_, 1);

  // QKV
  gemm(cvb, qT, qb, nullptr, qf, nullptr, (int)P, H_, H_, 0);
  gemm(cvb, kT, kb, nullptr, kf, nullptr, (int)P, H_, H_, 0);
  gemm(cvb, vT, vb, nullptr, vf, nullptr, (int)P, H_, H_, 0);

  // attention core -> ctx (bf16)
  k_attn<<<B_ * NH_, 32, 0, stream>>>(qf, kf, vf, cnm, ctxb);

  // attn output proj + residual(cv), then LN1
  gemm(ctxb, aoT, aob, cvf, ares, nullptr, (int)P, H_, H_, 0);
  k_ln<<<(int)P, 256, 0, stream>>>(ares, ln1g, ln1b, aof, aob16);

  // FFN: inter = gelu(attn_out @ int + b); fres = inter @ out + b + attn_out; LN2
  gemm(aob16, intT, intb, nullptr, nullptr, intb16, (int)P, I_, H_, 2);
  gemm(intb16, outT, outb, aof, fres, nullptr, (int)P, H_, I_, 0);
  k_ln<<<(int)P, 256, 0, stream>>>(fres, ln2g, ln2b, o2f, nullptr);

  // decoder + masked KL loss
  k_dec<<<(B_ * M_ * NL_ + 255) / 256, 256, 0, stream>>>(o2f, decw, decb, pred);
  k_loss<<<1, 256, 0, stream>>>(pred, tgt, cnm, (float*)d_out);
}
